// LocalRNN_69578470195782
// MI455X (gfx1250) — matrix operand order repaired and checked by reading, not compiled, not run on hardware
//
#include <hip/hip_runtime.h>
#include <hip/hip_bf16.h>
#include <cstdint>

// ---------------- types ----------------
typedef __attribute__((ext_vector_type(16))) __bf16 v16bf;
typedef __attribute__((ext_vector_type(8)))  __bf16 bf8;
typedef __attribute__((ext_vector_type(8)))  float  v8f;

// problem constants (b=8, l=2048, d=512, ksize=8)
#define BATCH 8
#define SEQL  2048
#define DIM   512
#define G3    1536      // 3*d
#define KS    8
#define NROWS (BATCH*SEQL)   // 16384
#define MB    32             // rows (sequences) per workgroup in recurrence
#define LDSP  (DIM + 8)      // padded LDS row stride (bf16 elems): 4-dword bank rotation

static __device__ __forceinline__ v16bf cat16(bf8 lo, bf8 hi) {
    return __builtin_shufflevector(lo, hi, 0,1,2,3,4,5,6,7,8,9,10,11,12,13,14,15);
}

static __device__ __forceinline__ v8f wmma_bf16(v16bf a, v16bf b, v8f c) {
    // (neg_a, A, neg_b, B, c_mod, C, reuse_a, reuse_b)
    return __builtin_amdgcn_wmma_f32_16x16x32_bf16(false, a, false, b, (short)0, c, false, false);
}

// B fragment (32x16, K x N) from row-major bf16 W (ld=512): B[k][n] = W[n][k].
// lanes 0-15: col n0+lane, K = k0+{0..7,16..23} ; lanes 16-31: col n0+lane-16, K = k0+{8..15,24..31}
static __device__ __forceinline__ v16bf loadB(const __bf16* __restrict__ W,
                                              int n0, int k0, int lane) {
    int n  = n0 + (lane & 15);
    int kk = k0 + ((lane < 16) ? 0 : 8);
    const bf8* p = (const bf8*)(W + (size_t)n * DIM + kk);
    return cat16(p[0], p[2]);
}

// A fragment (16x32, M x K) from row-major bf16 X (ld=512), global memory.
// lanes 0-15: row m0+lane, K = k0+{0..7,16..23}; lanes 16-31: row m0+lane-16, K = k0+{8..15,24..31}
static __device__ __forceinline__ v16bf loadA_g(const __bf16* __restrict__ X,
                                                int m0, int k0, int lane) {
    int r  = m0 + (lane & 15);
    int kb = k0 + ((lane < 16) ? 0 : 8);
    const __bf16* base = X + (size_t)r * DIM + kb;
    return cat16(*(const bf8*)base, *(const bf8*)(base + 16));
}

// A fragment from bf16 LDS h shadow (padded ld=LDSP): direct ds_load_b128, no cvt.
static __device__ __forceinline__ v16bf loadA_lds(const __bf16* hsb,
                                                  int mt, int k0, int lane) {
    int lr = mt * 16 + (lane & 15);
    int kb = k0 + ((lane < 16) ? 0 : 8);
    const __bf16* p = hsb + (size_t)lr * LDSP + kb;
    return cat16(*(const bf8*)p, *(const bf8*)(p + 16));
}

static __device__ __forceinline__ float sigm(float x) {
    return 1.0f / (1.0f + __expf(-x));
}

// ---------------- kernel 0: f32 -> bf16 convert ----------------
__global__ void cvt_f32_bf16(const float* __restrict__ in, __bf16* __restrict__ out, int n) {
    int i = blockIdx.x * blockDim.x + threadIdx.x;
    int stride = gridDim.x * blockDim.x;
    for (; i < n; i += stride) out[i] = (__bf16)in[i];
}

// ---------------- kernel 1: y = x @ W_ih^T (no bias), f32 out ----------------
// grid: 12288 blocks * 256 threads; each wave = one 16x16 tile; 1024 x 96 tiles.
__global__ __launch_bounds__(256, 1)
void gemm_y(const __bf16* __restrict__ xb, const __bf16* __restrict__ wih,
            float* __restrict__ y) {
    int lane = threadIdx.x & 31;
    int wave = threadIdx.x >> 5;
    int tile = blockIdx.x * 8 + wave;
    int m0 = (tile / (G3 / 16)) * 16;
    int n0 = (tile % (G3 / 16)) * 16;

    v8f acc = {};
    for (int k0 = 0; k0 < DIM; k0 += 32) {
        v16bf a = loadA_g(xb, m0, k0, lane);
        v16bf b = loadB(wih, n0, k0, lane);
        acc = wmma_bf16(a, b, acc);
    }
    int col = n0 + (lane & 15);
    int rb  = m0 + ((lane < 16) ? 0 : 8);
#pragma unroll
    for (int i = 0; i < 8; ++i)
        y[(size_t)(rb + i) * G3 + col] = acc[i];
}

// ---------------- kernel 2: 8-step GRU recurrence ----------------
// 512 blocks * 256 threads; block owns MB=32 sequences.
// f32 master h lives in registers (each wave owns the exact elements it updates);
// a bf16 shadow of h lives in LDS (33 KB, padded) purely as the WMMA A operand.
// Wave w handles triples t=w*8..w*8+7; triple -> (mt = t>>5, col tile ct = t&31),
// computing gate columns (j0, j0+512, j0+1024) for 16 rows.
__global__ __launch_bounds__(256, 1)
void gru_rec(const __bf16* __restrict__ whh, const float* __restrict__ y,
             const float* __restrict__ b_ih, const float* __restrict__ b_hh,
             float* __restrict__ out) {
    __shared__ __bf16 hsb[MB * LDSP];         // 33,280 B
    int tid  = threadIdx.x;
    int lane = tid & 31;
    int wave = tid >> 5;
    int row0 = blockIdx.x * MB;

    for (int i = tid; i < MB * LDSP; i += 256) hsb[i] = (__bf16)0.0f;

    float hreg[8][8];                         // f32 master copy of owned h elems
#pragma unroll
    for (int t = 0; t < 8; ++t)
#pragma unroll
        for (int i = 0; i < 8; ++i) hreg[t][i] = 0.0f;

    __syncthreads();

    for (int k = 0; k < KS; ++k) {
#pragma unroll
        for (int t = 0; t < 8; ++t) {
            int trip = wave * 8 + t;
            int mt = trip >> 5;               // 0..1 (16-row sub-block)
            int j0 = (trip & 31) * 16;        // h / n-gate column base
            int colL = lane & 15;
            int cr = j0 + colL, cz = cr + DIM, cn = cz + DIM;

            v8f ar = {}, az = {}, an = {};
            for (int k0 = 0; k0 < DIM; k0 += 32) {
                v16bf a  = loadA_lds(hsb, mt, k0, lane);
                v16bf br = loadB(whh, j0,           k0, lane);
                v16bf bz = loadB(whh, j0 + DIM,     k0, lane);
                v16bf bn = loadB(whh, j0 + 2 * DIM, k0, lane);
                ar = wmma_bf16(a, br, ar);
                az = wmma_bf16(a, bz, az);
                an = wmma_bf16(a, bn, an);
            }

            float bihr = b_ih[cr], bihz = b_ih[cz], bihn = b_ih[cn];
            float bhhr = b_hh[cr], bhhz = b_hh[cz], bhhn = b_hh[cn];
            int rbase = row0 + mt * 16 + ((lane < 16) ? 0 : 8);
#pragma unroll
            for (int i = 0; i < 8; ++i) {
                int row = rbase + i;
                int l   = row & (SEQL - 1);
                float ir = bihr, iz = bihz, in_ = bihn;
                if (l + k >= KS - 1) {        // window position in-bounds
                    const float* yr = y + (size_t)(row - (KS - 1) + k) * G3;
                    ir += yr[cr]; iz += yr[cz]; in_ += yr[cn];
                }
                float r  = sigm(ir + ar[i] + bhhr);
                float z  = sigm(iz + az[i] + bhhz);
                float nn = tanhf(in_ + r * (an[i] + bhhn));
                hreg[t][i] = (1.0f - z) * nn + z * hreg[t][i];
            }
        }
        __syncthreads();                       // all GEMM reads of old h done
#pragma unroll
        for (int t = 0; t < 8; ++t) {
            int trip = wave * 8 + t;
            int mt = trip >> 5;
            int j0 = (trip & 31) * 16;
#pragma unroll
            for (int i = 0; i < 8; ++i) {
                int lr = mt * 16 + i + ((lane < 16) ? 0 : 8);
                hsb[(size_t)lr * LDSP + j0 + (lane & 15)] = (__bf16)hreg[t][i];
            }
        }
        __syncthreads();                       // bf16 shadow updated for next step
    }

    // final h written at full f32 precision straight from registers
#pragma unroll
    for (int t = 0; t < 8; ++t) {
        int trip = wave * 8 + t;
        int mt = trip >> 5;
        int j0 = (trip & 31) * 16;
        int col = j0 + (lane & 15);
        int rbase = row0 + mt * 16 + ((lane < 16) ? 0 : 8);
#pragma unroll
        for (int i = 0; i < 8; ++i)
            out[(size_t)(rbase + i) * DIM + col] = hreg[t][i];
    }
}

// ---------------- launch ----------------
extern "C" void kernel_launch(void* const* d_in, const int* in_sizes, int n_in,
                              void* d_out, int out_size, void* d_ws, size_t ws_size,
                              hipStream_t stream) {
    const float* x    = (const float*)d_in[0];   // (8,2048,512)
    const float* W_ih = (const float*)d_in[1];   // (1536,512)
    const float* W_hh = (const float*)d_in[2];   // (1536,512)
    const float* b_ih = (const float*)d_in[3];   // (1536,)
    const float* b_hh = (const float*)d_in[4];   // (1536,)
    float* out = (float*)d_out;                  // (8,2048,512)

    char* ws = (char*)d_ws;
    // workspace layout (all 16B-aligned): xb | W_ih(bf16) | W_hh(bf16) | y(f32)
    const size_t NX = (size_t)NROWS * DIM;          // 8,388,608
    const size_t NW = (size_t)G3 * DIM;             // 786,432
    __bf16* xb   = (__bf16*)ws;
    __bf16* wihb = (__bf16*)(ws + NX * 2);
    __bf16* whhb = (__bf16*)(ws + NX * 2 + NW * 2);
    float*  yws  = (float*) (ws + NX * 2 + NW * 4); // 16384*1536 f32

    cvt_f32_bf16<<<2048, 256, 0, stream>>>(x,    xb,   (int)NX);
    cvt_f32_bf16<<<512,  256, 0, stream>>>(W_ih, wihb, (int)NW);
    cvt_f32_bf16<<<512,  256, 0, stream>>>(W_hh, whhb, (int)NW);

    // y = x @ W_ih^T : (16384/16)*(1536/16) = 98304 wave-tiles / 8 waves per block
    gemm_y<<<98304 / 8, 256, 0, stream>>>(xb, wihb, yws);

    // recurrence: 16384 rows / 32 per block
    gru_rec<<<NROWS / MB, 256, 0, stream>>>(whhb, yws, b_ih, b_hh, out);
}
